// ScaleDotProductAttention_26018911879443
// MI455X (gfx1250) — matrix operand order repaired and checked by reading, not compiled, not run on hardware
//
#include <hip/hip_runtime.h>
#include <hip/hip_bf16.h>
#include <cstdint>

// ---------------------------------------------------------------------------
// Attention: q=QWq^T+bq, k=KWk^T+bk, v=VWv^T+bv ; S=qk^T/32 (masked);
// alpha=softmax(S); out=alpha v.    B=8, L1=L2=2048, D=1024.
// bf16 WMMA (v_wmma_f32_16x16x32_bf16) for all GEMMs, f32 accumulate.
// Double-buffered LDS tiles; async global->LDS (ASYNCcnt) when available.
// ---------------------------------------------------------------------------

typedef unsigned short u16;
typedef __attribute__((ext_vector_type(16))) __bf16 v16bf;
typedef __attribute__((ext_vector_type(8)))  __bf16 v8bf;
typedef __attribute__((ext_vector_type(8)))  float  v8f;
typedef __attribute__((ext_vector_type(4)))  int    v4i;

#define BATCH 8
#define SEQ   2048
#define DIM   1024
#define BK    32
#define LDSW  40   // 32 + 8 pad (80B row stride: 16B aligned, bank-spread)

#if __has_builtin(__builtin_amdgcn_global_load_async_to_lds_b128)
#define HAVE_ASYNC_LDS 1
#else
#define HAVE_ASYNC_LDS 0
#endif

typedef __attribute__((address_space(1))) v4i* gas_v4i;
typedef __attribute__((address_space(3))) v4i* lds_v4i;

__device__ __forceinline__ gas_v4i to_global(const void* p) {
  return (gas_v4i)(uintptr_t)p;
}
__device__ __forceinline__ lds_v4i to_lds(void* p) {
  // low 32 bits of a generic LDS address are the LDS byte offset
  return (lds_v4i)(unsigned)(uintptr_t)p;
}
__device__ __forceinline__ void wait_async0() {
#if __has_builtin(__builtin_amdgcn_s_wait_asynccnt)
  __builtin_amdgcn_s_wait_asynccnt(0);
#else
  asm volatile("s_wait_asynccnt 0" ::: "memory");
#endif
}

union Frag { v16bf v; v8bf h[2]; };

__device__ __forceinline__ u16 f2bf(float f) {
  unsigned u = __float_as_uint(f);
  u += 0x7FFFu + ((u >> 16) & 1u);   // round-to-nearest-even
  return (u16)(u >> 16);
}

__device__ __forceinline__ v8f wmma_bf16(const Frag& a, const Frag& b, v8f c) {
  return __builtin_amdgcn_wmma_f32_16x16x32_bf16(false, a.v, false, b.v,
                                                 (short)0, c, false, false);
}

// A-fragment: lane holds row m=lane%16; K runs [koff,koff+8),[koff+16,koff+24)
__device__ __forceinline__ Frag load_fragA(const u16* rowp, int lane) {
  Frag f;
  const int koff = (lane >= 16) ? 8 : 0;
  f.h[0] = *(const v8bf*)(rowp + koff);
  f.h[1] = *(const v8bf*)(rowp + koff + 16);
  return f;
}
// B-fragment: lane holds col n=lane%16; K runs [koff,koff+8),[koff+16,koff+24)
__device__ __forceinline__ Frag load_fragB(const u16* rowp, int lane) {
  Frag f;
  const int koff = (lane >= 16) ? 8 : 0;
  f.h[0] = *(const v8bf*)(rowp + koff);
  f.h[1] = *(const v8bf*)(rowp + koff + 16);
  return f;
}

// ---------------------------------------------------------------------------
// Kernel 1: out_bf16[M,1024] = act_f32[M,1024] @ W_f32[1024,1024]^T + bias
// 128x128 tile / WG, 8 waves each 64x32 (4x2 WMMA tiles). Double-buffered:
// global loads for tile k+1 are in flight during WMMA on tile k.
// ---------------------------------------------------------------------------
__global__ __launch_bounds__(256) void proj_gemm_kernel(
    const float* __restrict__ A, const float* __restrict__ W,
    const float* __restrict__ bias, u16* __restrict__ out) {
  __shared__ u16 sA[2][128][LDSW];
  __shared__ u16 sB[2][128][LDSW];
  const int tid  = threadIdx.x;
  const int lane = tid & 31, wave = tid >> 5;
  const int wy = wave >> 2, wx = wave & 3;          // 2x4 wave grid
  const int m0 = blockIdx.x * 128, n0 = blockIdx.y * 128;

  v8f acc[4][2] = {};
  float4 ra[4], rb[4];

  auto gload = [&](int k0) {
#pragma unroll
    for (int r = 0; r < 4; ++r) {
      const int idx = tid + r * 256;          // float4 index 0..1023
      const int row = idx >> 3;
      const int c4  = (idx & 7) * 4;
      ra[r] = *(const float4*)(A + (size_t)(m0 + row) * DIM + k0 + c4);
      rb[r] = *(const float4*)(W + (size_t)(n0 + row) * DIM + k0 + c4);
    }
  };
  auto lstore = [&](int buf) {
#pragma unroll
    for (int r = 0; r < 4; ++r) {
      const int idx = tid + r * 256;
      const int row = idx >> 3;
      const int c4  = (idx & 7) * 4;
      ushort4 ha, hb;
      ha.x = f2bf(ra[r].x); ha.y = f2bf(ra[r].y);
      ha.z = f2bf(ra[r].z); ha.w = f2bf(ra[r].w);
      hb.x = f2bf(rb[r].x); hb.y = f2bf(rb[r].y);
      hb.z = f2bf(rb[r].z); hb.w = f2bf(rb[r].w);
      *(ushort4*)(&sA[buf][row][c4]) = ha;
      *(ushort4*)(&sB[buf][row][c4]) = hb;
    }
  };

  gload(0); lstore(0); __syncthreads();

  const int NIT = DIM / BK;
  for (int it = 0; it < NIT; ++it) {
    const int cur = it & 1;
    if (it + 1 < NIT) gload((it + 1) * BK);   // overlap with WMMA below

    Frag bf[2];
#pragma unroll
    for (int nt = 0; nt < 2; ++nt)
      bf[nt] = load_fragB(&sB[cur][wx * 32 + nt * 16 + (lane & 15)][0], lane);
#pragma unroll
    for (int mt = 0; mt < 4; ++mt) {
      Frag af = load_fragA(&sA[cur][wy * 64 + mt * 16 + (lane & 15)][0], lane);
#pragma unroll
      for (int nt = 0; nt < 2; ++nt)
        acc[mt][nt] = wmma_bf16(af, bf[nt], acc[mt][nt]);
    }

    if (it + 1 < NIT) lstore(cur ^ 1);
    __syncthreads();
  }

  // Epilogue: +bias, bf16. C/D layout: M=r+8*(lane>=16), N=lane%16.
  const int rsel = (lane >= 16) ? 8 : 0;
#pragma unroll
  for (int mt = 0; mt < 4; ++mt)
#pragma unroll
    for (int nt = 0; nt < 2; ++nt) {
      const int col = n0 + wx * 32 + nt * 16 + (lane & 15);
      const float bc = bias[col];
#pragma unroll
      for (int r = 0; r < 8; ++r) {
        const int row = m0 + wy * 64 + mt * 16 + rsel + r;
        out[(size_t)row * DIM + col] = f2bf(acc[mt][nt][r] + bc);
      }
    }
}

// ---------------------------------------------------------------------------
// Kernel 2: S[b,i,j] = (q[b,i,:].k[b,j,:]) / 32, masked with -inf.
// bf16->bf16 staging: async global->LDS (ASYNCcnt) when available.
// ---------------------------------------------------------------------------
__global__ __launch_bounds__(256) void scores_gemm_kernel(
    const u16* __restrict__ qb, const u16* __restrict__ kb,
    const int* __restrict__ kmask, float* __restrict__ S) {
  __shared__ u16 sA[2][128][LDSW];
  __shared__ u16 sB[2][128][LDSW];
  const int tid  = threadIdx.x;
  const int lane = tid & 31, wave = tid >> 5;
  const int wy = wave >> 2, wx = wave & 3;
  const int b  = blockIdx.z;
  const int m0 = blockIdx.x * 128, n0 = blockIdx.y * 128;
  const u16* Aq = qb + (size_t)b * SEQ * DIM;
  const u16* Bk = kb + (size_t)b * SEQ * DIM;

  v8f acc[4][2] = {};

#if HAVE_ASYNC_LDS
  auto stage = [&](int buf, int k0) {
#pragma unroll
    for (int r = 0; r < 2; ++r) {
      const int idx = tid + r * 256;          // 16B chunk index 0..511
      const int row = idx >> 2;
      const int c8  = (idx & 3) * 8;
      __builtin_amdgcn_global_load_async_to_lds_b128(
          to_global(Aq + (size_t)(m0 + row) * DIM + k0 + c8),
          to_lds(&sA[buf][row][c8]), 0, 0);
      __builtin_amdgcn_global_load_async_to_lds_b128(
          to_global(Bk + (size_t)(n0 + row) * DIM + k0 + c8),
          to_lds(&sB[buf][row][c8]), 0, 0);
    }
  };
  stage(0, 0); wait_async0(); __syncthreads();
#else
  uint4 rA[2], rB[2];
  auto gload = [&](int k0) {
#pragma unroll
    for (int r = 0; r < 2; ++r) {
      const int idx = tid + r * 256;
      const int row = idx >> 2;
      const int c8  = (idx & 3) * 8;
      rA[r] = *(const uint4*)(Aq + (size_t)(m0 + row) * DIM + k0 + c8);
      rB[r] = *(const uint4*)(Bk + (size_t)(n0 + row) * DIM + k0 + c8);
    }
  };
  auto lstore = [&](int buf) {
#pragma unroll
    for (int r = 0; r < 2; ++r) {
      const int idx = tid + r * 256;
      const int row = idx >> 2;
      const int c8  = (idx & 3) * 8;
      *(uint4*)(&sA[buf][row][c8]) = rA[r];
      *(uint4*)(&sB[buf][row][c8]) = rB[r];
    }
  };
  gload(0); lstore(0); __syncthreads();
#endif

  const int NIT = DIM / BK;
  for (int it = 0; it < NIT; ++it) {
    const int cur = it & 1;
#if HAVE_ASYNC_LDS
    if (it + 1 < NIT) stage(cur ^ 1, (it + 1) * BK);   // streams during WMMA
#else
    if (it + 1 < NIT) gload((it + 1) * BK);
#endif

    Frag bf[2];
#pragma unroll
    for (int nt = 0; nt < 2; ++nt)
      bf[nt] = load_fragB(&sB[cur][wx * 32 + nt * 16 + (lane & 15)][0], lane);
#pragma unroll
    for (int mt = 0; mt < 4; ++mt) {
      Frag af = load_fragA(&sA[cur][wy * 64 + mt * 16 + (lane & 15)][0], lane);
#pragma unroll
      for (int nt = 0; nt < 2; ++nt)
        acc[mt][nt] = wmma_bf16(af, bf[nt], acc[mt][nt]);
    }

#if HAVE_ASYNC_LDS
    if (it + 1 < NIT) wait_async0();
#else
    if (it + 1 < NIT) lstore(cur ^ 1);
#endif
    __syncthreads();
  }

  const float scale = 0.03125f;   // 1/sqrt(1024)
  const int rsel = (lane >= 16) ? 8 : 0;
#pragma unroll
  for (int mt = 0; mt < 4; ++mt)
#pragma unroll
    for (int nt = 0; nt < 2; ++nt) {
      const int col = n0 + wx * 32 + nt * 16 + (lane & 15);
      const int mk  = kmask[b * SEQ + col];
#pragma unroll
      for (int r = 0; r < 8; ++r) {
        const int row = m0 + wy * 64 + mt * 16 + rsel + r;
        const float val = mk ? -__builtin_inff() : acc[mt][nt][r] * scale;
        S[((size_t)b * SEQ + row) * SEQ + col] = val;
      }
    }
}

// ---------------------------------------------------------------------------
// Kernel 3: row softmax over 2048 f32 -> alpha bf16. One WG per row.
// ---------------------------------------------------------------------------
__global__ __launch_bounds__(256) void softmax_kernel(
    const float* __restrict__ S, u16* __restrict__ alpha) {
  __shared__ float red[256];
  const int row = blockIdx.x;
  const int tid = threadIdx.x;
  const float* src = S + (size_t)row * SEQ;

  float x[8];
  float m = -__builtin_inff();
#pragma unroll
  for (int i = 0; i < 8; ++i) { x[i] = src[tid + i * 256]; m = fmaxf(m, x[i]); }
  red[tid] = m; __syncthreads();
  for (int s = 128; s > 0; s >>= 1) {
    if (tid < s) red[tid] = fmaxf(red[tid], red[tid + s]);
    __syncthreads();
  }
  m = red[0]; __syncthreads();

  float sum = 0.f;
#pragma unroll
  for (int i = 0; i < 8; ++i) { x[i] = __expf(x[i] - m); sum += x[i]; }
  red[tid] = sum; __syncthreads();
  for (int s = 128; s > 0; s >>= 1) {
    if (tid < s) red[tid] += red[tid + s];
    __syncthreads();
  }
  const float inv = 1.0f / red[0];

  u16* dst = alpha + (size_t)row * SEQ;
#pragma unroll
  for (int i = 0; i < 8; ++i) dst[tid + i * 256] = f2bf(x[i] * inv);
}

// ---------------------------------------------------------------------------
// Kernel 4: out[b,i,d] = sum_j alpha[b,i,j] * v[b,j,d]   (f32 output)
// alpha tile: async->LDS when available. V tile: transposed through VGPRs
// into LDS so B-fragments see K(j)-contiguous runs; double-buffered.
// ---------------------------------------------------------------------------
__global__ __launch_bounds__(256) void av_gemm_kernel(
    const u16* __restrict__ alpha, const u16* __restrict__ vb,
    float* __restrict__ out) {
  __shared__ u16 sA[2][128][LDSW];
  __shared__ u16 sBT[2][128][LDSW];   // [d][j]
  const int tid  = threadIdx.x;
  const int lane = tid & 31, wave = tid >> 5;
  const int wy = wave >> 2, wx = wave & 3;
  const int b  = blockIdx.z;
  const int m0 = blockIdx.x * 128;   // query rows
  const int n0 = blockIdx.y * 128;   // output dims
  const u16* Aa = alpha + (size_t)b * SEQ * SEQ;
  const u16* Vv = vb    + (size_t)b * SEQ * DIM;

  v8f acc[4][2] = {};
  union U4 { uint4 u; u16 s[8]; };
  U4 tv[2];

#if HAVE_ASYNC_LDS
  auto stageA = [&](int buf, int k0) {
#pragma unroll
    for (int r = 0; r < 2; ++r) {
      const int idx = tid + r * 256;
      const int row = idx >> 2;
      const int c8  = (idx & 3) * 8;
      __builtin_amdgcn_global_load_async_to_lds_b128(
          to_global(Aa + (size_t)(m0 + row) * SEQ + k0 + c8),
          to_lds(&sA[buf][row][c8]), 0, 0);
    }
  };
#else
  uint4 rA[2];
  auto gloadA = [&](int k0) {
#pragma unroll
    for (int r = 0; r < 2; ++r) {
      const int idx = tid + r * 256;
      const int row = idx >> 2;
      const int c8  = (idx & 3) * 8;
      rA[r] = *(const uint4*)(Aa + (size_t)(m0 + row) * SEQ + k0 + c8);
    }
  };
  auto lstoreA = [&](int buf) {
#pragma unroll
    for (int r = 0; r < 2; ++r) {
      const int idx = tid + r * 256;
      const int row = idx >> 2;
      const int c8  = (idx & 3) * 8;
      *(uint4*)(&sA[buf][row][c8]) = rA[r];
    }
  };
#endif
  auto gloadV = [&](int k0) {
#pragma unroll
    for (int r = 0; r < 2; ++r) {
      const int idx = tid + r * 256;       // 16 uint4 per j-row
      const int jj  = idx >> 4;            // 0..31
      const int c8  = (idx & 15) * 8;      // d offset
      tv[r].u = *(const uint4*)(Vv + (size_t)(k0 + jj) * DIM + n0 + c8);
    }
  };
  auto lstoreV = [&](int buf) {            // transpose into sBT[d][j]
#pragma unroll
    for (int r = 0; r < 2; ++r) {
      const int idx = tid + r * 256;
      const int jj  = idx >> 4;
      const int c8  = (idx & 15) * 8;
#pragma unroll
      for (int e = 0; e < 8; ++e) sBT[buf][c8 + e][jj] = tv[r].s[e];
    }
  };

#if HAVE_ASYNC_LDS
  stageA(0, 0); gloadV(0); lstoreV(0); wait_async0(); __syncthreads();
#else
  gloadA(0); gloadV(0); lstoreA(0); lstoreV(0); __syncthreads();
#endif

  const int NIT = SEQ / BK;
  for (int it = 0; it < NIT; ++it) {
    const int cur = it & 1;
    if (it + 1 < NIT) {
#if HAVE_ASYNC_LDS
      stageA(cur ^ 1, (it + 1) * BK);
#else
      gloadA((it + 1) * BK);
#endif
      gloadV((it + 1) * BK);
    }

    Frag bf[2];
#pragma unroll
    for (int nt = 0; nt < 2; ++nt)
      bf[nt] = load_fragB(&sBT[cur][wx * 32 + nt * 16 + (lane & 15)][0], lane);
#pragma unroll
    for (int mt = 0; mt < 4; ++mt) {
      Frag af = load_fragA(&sA[cur][wy * 64 + mt * 16 + (lane & 15)][0], lane);
#pragma unroll
      for (int nt = 0; nt < 2; ++nt)
        acc[mt][nt] = wmma_bf16(af, bf[nt], acc[mt][nt]);
    }

    if (it + 1 < NIT) {
      lstoreV(cur ^ 1);
#if HAVE_ASYNC_LDS
      wait_async0();
#else
      lstoreA(cur ^ 1);
#endif
    }
    __syncthreads();
  }

  const int rsel = (lane >= 16) ? 8 : 0;
#pragma unroll
  for (int mt = 0; mt < 4; ++mt)
#pragma unroll
    for (int nt = 0; nt < 2; ++nt) {
      const int col = n0 + wx * 32 + nt * 16 + (lane & 15);
#pragma unroll
      for (int r = 0; r < 8; ++r) {
        const int row = m0 + wy * 64 + mt * 16 + rsel + r;
        out[((size_t)b * SEQ + row) * DIM + col] = acc[mt][nt][r];
      }
    }
}

// ---------------------------------------------------------------------------
extern "C" void kernel_launch(void* const* d_in, const int* in_sizes, int n_in,
                              void* d_out, int out_size, void* d_ws, size_t ws_size,
                              hipStream_t stream) {
  const float* Q     = (const float*)d_in[0];
  const float* K     = (const float*)d_in[1];
  const float* V     = (const float*)d_in[2];
  const int*   Kmask = (const int*)  d_in[3];
  const float* Wq    = (const float*)d_in[4];
  const float* bq    = (const float*)d_in[5];
  const float* Wk    = (const float*)d_in[6];
  const float* bk    = (const float*)d_in[7];
  const float* Wv    = (const float*)d_in[8];
  const float* bv    = (const float*)d_in[9];
  float* out = (float*)d_out;

  // Workspace layout
  char* ws = (char*)d_ws;
  const size_t nProj = (size_t)BATCH * SEQ * DIM;       // 16.7M elems
  const size_t nScor = (size_t)BATCH * SEQ * SEQ;       // 33.6M elems
  u16*   qb    = (u16*)ws;
  u16*   kb    = qb + nProj;
  u16*   vbuf  = kb + nProj;
  float* S     = (float*)(ws + 3 * nProj * sizeof(u16));
  u16*   alpha = (u16*)(ws + 3 * nProj * sizeof(u16) + nScor * sizeof(float));

  const dim3 blk(256);

  // 1) Projections: 16384x1024 @ 1024x1024^T + bias
  const dim3 gProj((BATCH * SEQ) / 128, DIM / 128);
  proj_gemm_kernel<<<gProj, blk, 0, stream>>>(Q, Wq, bq, qb);
  proj_gemm_kernel<<<gProj, blk, 0, stream>>>(K, Wk, bk, kb);
  proj_gemm_kernel<<<gProj, blk, 0, stream>>>(V, Wv, bv, vbuf);

  // 2) Masked scaled scores
  const dim3 gS(SEQ / 128, SEQ / 128, BATCH);
  scores_gemm_kernel<<<gS, blk, 0, stream>>>(qb, kb, Kmask, S);

  // 3) Softmax -> bf16 alpha
  softmax_kernel<<<dim3(BATCH * SEQ), blk, 0, stream>>>(S, alpha);

  // 4) alpha @ v -> f32 output
  const dim3 gAV(SEQ / 128, DIM / 128, BATCH);
  av_gemm_kernel<<<gAV, blk, 0, stream>>>(alpha, vbuf, out);
}
